// Kernel_4262547987266090318_10986526343736
// MI455X (gfx1250) — matrix operand order repaired and checked by reading, not compiled, not run on hardware
//
#include <hip/hip_runtime.h>
#include <hip/hip_bf16.h>
#include <stdint.h>

// ---------------------------------------------------------------------------
// Shapes (compile-time constants from the reference):
//   x  : [8,128,56,56] f32      w3 : [16,128,5,7] f32
//   w4 : [16,16] f32            w5 : [128,16] f32
//   out: [8,128,56,56] f32
// Fusion: out = (w5 @ w4) @ dilated_conv(x, w3)
// Conv runs as implicit GEMM on v_wmma_f32_16x16x32_f16 with f16 hi/lo
// split-precision (fp32-grade accuracy, ~3-4x faster than the fp32 WMMA path).
// ---------------------------------------------------------------------------

typedef __attribute__((ext_vector_type(16))) _Float16 v16h;
typedef __attribute__((ext_vector_type(8)))  float    v8f;

#define NIMG 8
#define C_IN 128
#define HH 56
#define WW 56
#define SPAT (HH * WW)          // 3136
#define NPOS (NIMG * SPAT)      // 25088
#define CO 16
#define NTAP 35                 // 5x7 taps
#define NCC 4                   // 128 channels / 32 per WMMA K
#define NTILE (NIMG * HH * 4)   // 1792 output tiles (w starts {0,16,32,40})

// ---------------- Kernel 1a: split x into packed f16 hi|lo ----------------
__global__ __launch_bounds__(256) void pack_x_kernel(const float* __restrict__ x,
                                                     unsigned* __restrict__ xpk) {
    const int i = blockIdx.x * 256 + threadIdx.x;   // exactly 3,211,264 threads
    const float v = x[i];
    const _Float16 hi = (_Float16)v;
    const _Float16 lo = (_Float16)(v - (float)hi);
    const unsigned uh = (unsigned)__builtin_bit_cast(unsigned short, hi);
    const unsigned ul = (unsigned)__builtin_bit_cast(unsigned short, lo);
    xpk[i] = uh | (ul << 16);
}

// ---- Kernel 1b: split w3 into f16 hi/lo in exact WMMA A per-lane order ----
// Layout: wp[((tap*4 + cc)*32 + lane)*16 + e]
// A 16x32 f16 map: M = lane%16, K(lane,e) = e + (e&8) + 8*(lane>=16)
__global__ __launch_bounds__(256) void pack_w_kernel(const float* __restrict__ w3,
                                                     _Float16* __restrict__ wphi,
                                                     _Float16* __restrict__ wplo) {
    const int t = blockIdx.x * 256 + threadIdx.x;   // 35*4*32*16 = 71,680 threads
    const int e    = t & 15;
    const int lane = (t >> 4) & 31;
    const int cc   = (t >> 9) & 3;
    const int tap  = t >> 11;
    const int co = lane & 15;
    const int kl = (e + (e & 8)) + ((lane & 16) >> 1);   // hardware K slot 0..31
    const int c  = cc * 32 + kl;
    const int kh = tap / 7, kw = tap % 7;
    const float v = w3[((co * C_IN + c) * 5 + kh) * 7 + kw];
    const _Float16 hi = (_Float16)v;
    wphi[t] = hi;
    wplo[t] = (_Float16)(v - (float)hi);
}

// ---------------- Kernel 1c: Wc = w5 @ w4  (128x16) ----------------
__global__ __launch_bounds__(256) void wc_kernel(const float* __restrict__ w4,
                                                 const float* __restrict__ w5,
                                                 float* __restrict__ Wc) {
    const int t = blockIdx.x * 256 + threadIdx.x;   // 2048 threads
    if (t >= C_IN * CO) return;
    const int o = t >> 4, co = t & 15;
    float s = 0.f;
#pragma unroll
    for (int k = 0; k < 16; ++k) s = fmaf(w5[o * 16 + k], w4[k * 16 + co], s);
    Wc[t] = s;
}

// ---------------- Kernel 2: dilated conv as implicit GEMM on WMMA ----------
// One wave per (tile, K-split): D[16 co x 16 w-positions].
// blockIdx.y in {0,1} selects channel chunks {0,1} or {2,3} (balanced K split
// -> 3584 waves for latency hiding); partials summed in the epilogue.
// Zero padding via unconditional loads + value masking (guard zones around xpk
// make OOB-lane addresses safe), so EXEC stays all-1s: no mask churn, and all
// kw/e offsets fold into the 24-bit global_load immediate.
__global__ __launch_bounds__(256) void conv_wmma_kernel(const unsigned* __restrict__ xpk,
                                                        const _Float16* __restrict__ wphi,
                                                        const _Float16* __restrict__ wplo,
                                                        float* __restrict__ y1) {
    const int lane = threadIdx.x;                       // 0..31 (wave32)
    const int tile = blockIdx.x * 8 + threadIdx.y;      // 0..1791
    const int ks   = blockIdx.y;                        // K split 0/1
    int t = tile;
    const int wc   = t & 3;  t >>= 2;
    const int h    = t % HH;
    const int nImg = t / HH;
    const int w0   = (wc < 3) ? wc * 16 : 40;           // overlap at 40..47 is benign (same data)

    const int p       = lane & 15;                      // B column = w position in tile
    const int hiHalf  = lane >> 4;                      // 0/1: K half for B, M half for D
    const int cOff    = hiHalf << 3;                    // +8 channels for upper-lane K half
    const int imgBase = nImg * (C_IN * SPAT);

    // per-lane OOB masks for the 7 kw taps (iw = w0 + p + 3*kw - 9)
    unsigned msk[7];
#pragma unroll
    for (int kw = 0; kw < 7; ++kw) {
        const int iw = w0 + p + 3 * kw - 9;
        msk[kw] = ((iw >= 0) & (iw < WW)) ? 0xffffffffu : 0u;
    }

    v8f acc = {0.f, 0.f, 0.f, 0.f, 0.f, 0.f, 0.f, 0.f};

    for (int kh = 0; kh < 5; ++kh) {
        const int ih = h + 2 * kh - 4;
        if (ih < 0 || ih >= HH) continue;               // uniform: whole row padded out
        const int pbase = imgBase + ih * WW + (w0 + p - 9);
#pragma unroll
        for (int cci = 0; cci < 2; ++cci) {
            const int cc = 2 * ks + cci;
            const unsigned* src = xpk + pbase + (cc * 32 + cOff) * SPAT;   // one base per (kh,cc)
            const _Float16* wbase_h = wphi + ((kh * 7 * NCC + cc) * 32 + lane) * 16;
            const _Float16* wbase_l = wplo + ((kh * 7 * NCC + cc) * 32 + lane) * 16;
#pragma unroll
            for (int kw = 0; kw < 7; ++kw) {
                v16h ah = *(const v16h*)(wbase_h + kw * (NCC * 32 * 16)); // const-offset b128 x2
                v16h al = *(const v16h*)(wbase_l + kw * (NCC * 32 * 16));
                const unsigned m = msk[kw];
                v16h bh, bl;
#pragma unroll
                for (int e = 0; e < 16; ++e) {          // const-offset b32 gathers, 64B coalesced
                    const unsigned u = src[(e + (e & 8)) * SPAT + 3 * kw] & m;
                    bh[e] = __builtin_bit_cast(_Float16, (unsigned short)(u & 0xffffu));
                    bl[e] = __builtin_bit_cast(_Float16, (unsigned short)(u >> 16));
                }
                acc = __builtin_amdgcn_wmma_f32_16x16x32_f16(false, ah, false, bh, (short)0, acc, false, false);
                acc = __builtin_amdgcn_wmma_f32_16x16x32_f16(false, ah, false, bl, (short)0, acc, false, false);
                acc = __builtin_amdgcn_wmma_f32_16x16x32_f16(false, al, false, bh, (short)0, acc, false, false);
            }
        }
    }

    // D layout: lane holds position p, VGPR r -> co = r + 8*hiHalf. Store y1[pos][16].
    const int posBase = (nImg * HH + h) * WW + w0;
    float* dst = y1 + (size_t)ks * (NPOS * CO) + (size_t)(posBase + p) * CO + hiHalf * 8;
    float4 v0 = {acc[0], acc[1], acc[2], acc[3]};
    float4 v1 = {acc[4], acc[5], acc[6], acc[7]};
    *(float4*)dst       = v0;
    *(float4*)(dst + 4) = v1;
}

// ---------------- Kernel 3: out = Wc @ (y1a + y1b) (memory-bound epilogue) --
__global__ __launch_bounds__(256) void proj_kernel(const float* __restrict__ y1,
                                                   const float* __restrict__ Wc,
                                                   float* __restrict__ out) {
    const int pos = blockIdx.x * 256 + threadIdx.x;     // 25088 = 98 * 256 exactly
    const int og  = blockIdx.y << 4;                    // 8 groups of 16 output channels
    const float4* ya = (const float4*)(y1 + (size_t)pos * CO);
    const float4* yb = (const float4*)(y1 + (size_t)NPOS * CO + (size_t)pos * CO);
    float yv[16];
#pragma unroll
    for (int q = 0; q < 4; ++q) {
        const float4 a = ya[q], b = yb[q];
        yv[4 * q + 0] = a.x + b.x;
        yv[4 * q + 1] = a.y + b.y;
        yv[4 * q + 2] = a.z + b.z;
        yv[4 * q + 3] = a.w + b.w;
    }
    const int n  = pos / SPAT;
    const int hw = pos - n * SPAT;
    float* obase = out + (size_t)n * (C_IN * SPAT) + hw;
#pragma unroll
    for (int oi = 0; oi < 16; ++oi) {
        const float* wr = Wc + (og + oi) * 16;          // uniform across wave -> scalar loads
        float s = 0.f;
#pragma unroll
        for (int k = 0; k < 16; ++k) s = fmaf(wr[k], yv[k], s);
        obase[(size_t)(og + oi) * SPAT] = s;            // coalesced along hw
    }
}

// ---------------------------------------------------------------------------
extern "C" void kernel_launch(void* const* d_in, const int* in_sizes, int n_in,
                              void* d_out, int out_size, void* d_ws, size_t ws_size,
                              hipStream_t stream) {
    (void)in_sizes; (void)n_in; (void)out_size; (void)ws_size;
    const float* x  = (const float*)d_in[0];
    const float* w3 = (const float*)d_in[1];
    const float* w4 = (const float*)d_in[2];
    const float* w5 = (const float*)d_in[3];
    float* out = (float*)d_out;

    // Workspace layout (256B-aligned, with 256B guard zones around xpk so
    // masked OOB-lane gathers (max +-36B) never leave the allocation):
    char* base = (char*)d_ws;
    unsigned* xpk  = (unsigned*)(base + 256);                    // 3,211,264 * 4 = 12,845,056 B
    _Float16* wphi = (_Float16*)(base + 256 + 12845056 + 256);   // 71,680 * 2    =    143,360 B
    _Float16* wplo = (_Float16*)(base + 256 + 12845056 + 256 + 143360);
    float*    Wc   = (float*)(base + 256 + 12845056 + 256 + 2 * 143360);        // 8,192 B
    float*    y1   = (float*)(base + 256 + 12845056 + 256 + 2 * 143360 + 8192); // 2 * 1,605,632 B
    // total ~15.6 MB

    pack_x_kernel<<<NPOS * C_IN / 256, 256, 0, stream>>>(x, xpk);                  // 12544 blocks
    pack_w_kernel<<<NTAP * NCC * 32 * 16 / 256, 256, 0, stream>>>(w3, wphi, wplo); // 280 blocks
    wc_kernel<<<(C_IN * CO + 255) / 256, 256, 0, stream>>>(w4, w5, Wc);            // 8 blocks
    conv_wmma_kernel<<<dim3(NTILE / 8, 2), dim3(32, 8), 0, stream>>>(xpk, wphi, wplo, y1); // 224x2
    proj_kernel<<<dim3(NPOS / 256, 8), 256, 0, stream>>>(y1, Wc, out);             // 98x8 blocks
}
